// EmbeddingGNN_11141145166539
// MI455X (gfx1250) — compile-verified, matrix operand order checked by reading, not run on hardware
//
#include <hip/hip_runtime.h>
#include <math.h>

typedef __bf16 bf16_t;
typedef __attribute__((ext_vector_type(16))) __bf16 v16bf;
typedef __attribute__((ext_vector_type(8)))  __bf16 v8bf;
typedef __attribute__((ext_vector_type(8)))  float  v8f;

// ---------------------------------------------------------------------------
// helpers
// ---------------------------------------------------------------------------
__device__ __forceinline__ float gelu_tanh(float x) {
    // jax.nn.gelu default (approximate=True)
    const float c0 = 0.7978845608028654f;  // sqrt(2/pi)
    float x3 = x * x * x;
    float t  = tanhf(c0 * (x + 0.044715f * x3));
    return 0.5f * x * (1.0f + t);
}

// Load one 16x32 bf16 A/B fragment (row-major source, row stride `stride`
// elements) for WMMA. Lane l<16 holds row (l&15), K = {0..7, 16..23} of the
// 32-wide K chunk; lane l>=16 holds K = {8..15, 24..31}. Both halves are
// contiguous 16-byte runs -> two b128 loads.
__device__ __forceinline__ v16bf load_frag(const bf16_t* __restrict__ base,
                                           int stride, int kchunk) {
    int lane = threadIdx.x & 31;
    int r    = lane & 15;
    int kb   = (lane & 16) ? 8 : 0;
    const bf16_t* p = base + (size_t)r * stride + kchunk * 32 + kb;
    v8bf lo = *(const v8bf*)p;
    v8bf hi = *(const v8bf*)(p + 16);
    v16bf out;
#pragma unroll
    for (int i = 0; i < 8; ++i) { out[i] = lo[i]; out[i + 8] = hi[i]; }
    return out;
}

// ---------------------------------------------------------------------------
// prep kernels
// ---------------------------------------------------------------------------
__global__ void transpose_to_bf16(const float* __restrict__ W,
                                  bf16_t* __restrict__ Wt, int K, int N) {
    int idx = blockIdx.x * blockDim.x + threadIdx.x;
    if (idx < K * N) {
        int k = idx / N, n = idx % N;
        Wt[(size_t)n * K + k] = (bf16_t)W[idx];
    }
}

__global__ void deg_init(float* deg, int n) {
    int i = blockIdx.x * blockDim.x + threadIdx.x;
    if (i < n) deg[i] = 1.0f;  // self-loop
}

__global__ void deg_accum(const int* __restrict__ col, float* __restrict__ deg, int E) {
    int e = blockIdx.x * blockDim.x + threadIdx.x;
    if (e < E) atomicAdd(&deg[col[e]], 1.0f);
}

__global__ void deg_finalize(float* deg, int n) {
    int i = blockIdx.x * blockDim.x + threadIdx.x;
    if (i < n) deg[i] = rsqrtf(deg[i]);  // deg -> deg^-1/2 in place
}

// h0 = concat(x[:, :64], emb[node_id])  ->  bf16 [N,128]
__global__ void build_h0(const float* __restrict__ x, const float* __restrict__ emb,
                         bf16_t* __restrict__ Hbf, int n) {
    int t = blockIdx.x * blockDim.x + threadIdx.x;
    if (t >= n * 128) return;
    int i = t >> 7, j = t & 127;
    float v;
    if (j < 64) {
        v = x[(size_t)i * 65 + j];
    } else {
        int nid = (int)x[(size_t)i * 65 + 64];
        v = emb[(size_t)nid * 64 + (j - 64)];
    }
    Hbf[t] = (bf16_t)v;
}

// ---------------------------------------------------------------------------
// GCN layer GEMM: Hc = H(bf16)[N,128] @ W[128,128]; also seeds the
// aggregation buffer with the self-loop term: agg = dis^2 * Hc + bias.
// Block = 16 rows; 8 waves, each owns one 16-wide column tile.
// ---------------------------------------------------------------------------
__global__ __launch_bounds__(256) void
gcn_gemm(const bf16_t* __restrict__ H, const bf16_t* __restrict__ Wt,
         const float* __restrict__ bias, const float* __restrict__ dis,
         float* __restrict__ Hc, float* __restrict__ agg) {
    __shared__ float sdis[16];
    int row0 = blockIdx.x * 16;
    int w    = threadIdx.x >> 5;   // wave id -> column tile
    int lane = threadIdx.x & 31;

    if (threadIdx.x < 16) sdis[threadIdx.x] = dis[row0 + threadIdx.x];
    __syncthreads();

    const bf16_t* Abase = H  + (size_t)row0 * 128;
    const bf16_t* Bbase = Wt + (size_t)w * 16 * 128;   // rows = output cols

    v8f acc = {};
#pragma unroll
    for (int kc = 0; kc < 4; ++kc) {
        v16bf a = load_frag(Abase, 128, kc);
        v16bf b = load_frag(Bbase, 128, kc);
        acc = __builtin_amdgcn_wmma_f32_16x16x32_bf16(
            false, a, false, b, (short)0, acc, false, false);
    }

    int nloc = lane & 15;
    int colg = w * 16 + nloc;
    int moff = (lane & 16) ? 8 : 0;
    float bc = bias[colg];
#pragma unroll
    for (int r = 0; r < 8; ++r) {
        int m = r + moff;
        size_t off = (size_t)(row0 + m) * 128 + colg;
        float v = acc[r];
        Hc[off]  = v;
        float d  = sdis[m];
        agg[off] = d * d * v + bc;
    }
}

// agg[col[e]] += dis[row]*dis[col] * Hc[row[e]]   (32 lanes / edge, float4)
__global__ __launch_bounds__(256) void
gcn_scatter(const int* __restrict__ row, const int* __restrict__ col,
            const float* __restrict__ dis, const float* __restrict__ Hc,
            float* __restrict__ agg, int E) {
    int t = blockIdx.x * blockDim.x + threadIdx.x;
    int e = t >> 5;
    if (e >= E) return;
    int j = (t & 31) * 4;
    int r = row[e], c = col[e];
    float nrm = dis[r] * dis[c];
    float4 hv = *(const float4*)(Hc + (size_t)r * 128 + j);
    float* dst = agg + (size_t)c * 128 + j;
    atomicAdd(dst + 0, nrm * hv.x);
    atomicAdd(dst + 1, nrm * hv.y);
    atomicAdd(dst + 2, nrm * hv.z);
    atomicAdd(dst + 3, nrm * hv.w);
}

// h_next = bf16(gelu(agg))
__global__ void gelu_to_bf16(const float* __restrict__ agg,
                             bf16_t* __restrict__ Hbf, int total) {
    int t = blockIdx.x * blockDim.x + threadIdx.x;
    if (t < total) Hbf[t] = (bf16_t)gelu_tanh(agg[t]);
}

// ---------------------------------------------------------------------------
// Edge MLP: out[e] = gelu(concat(h[row[e]], h[col[e]]) @ fcW + fcb)
// Block = 16 edges staged into padded LDS; 4 waves x (64/16) column tiles;
// K = 256 -> 8 bf16 WMMAs per wave.
// ---------------------------------------------------------------------------
#define LDSA_STRIDE 264   // 256 + 8 halves of pad: breaks LDS bank conflicts

__global__ __launch_bounds__(128) void
edge_mlp(const int* __restrict__ row, const int* __restrict__ col,
         const bf16_t* __restrict__ H, const bf16_t* __restrict__ fcWt,
         const float* __restrict__ fcb, float* __restrict__ out, int E) {
    __shared__ bf16_t sA[16 * LDSA_STRIDE];
    int e0  = blockIdx.x * 16;
    int tid = threadIdx.x;

    // Stage A tile: 16 edges x 256 bf16 (h[row] | h[col]); 8 threads / edge.
    {
        int m = tid >> 3;          // edge within tile
        int p = tid & 7;           // 8 x 64B segments
        int e = e0 + m;
        int node = (p < 4) ? row[e] : col[e];
        const bf16_t* src = H + (size_t)node * 128 + (p & 3) * 32;
        bf16_t* dst = sA + m * LDSA_STRIDE + p * 32;
#pragma unroll
        for (int q = 0; q < 4; ++q)
            *(v8bf*)(dst + q * 8) = *(const v8bf*)(src + q * 8);
    }
    __syncthreads();

    int w    = tid >> 5;           // wave -> output column tile (0..3)
    int lane = tid & 31;
    const bf16_t* Bbase = fcWt + (size_t)w * 16 * 256;

    v8f acc = {};
#pragma unroll
    for (int kc = 0; kc < 8; ++kc) {
        // A fragment from LDS (ds_load_b128 x2 per lane)
        int mm = lane & 15;
        int kb = (lane & 16) ? 8 : 0;
        const bf16_t* ap = sA + mm * LDSA_STRIDE + kc * 32 + kb;
        v8bf lo = *(const v8bf*)ap;
        v8bf hi = *(const v8bf*)(ap + 16);
        v16bf a;
#pragma unroll
        for (int i = 0; i < 8; ++i) { a[i] = lo[i]; a[i + 8] = hi[i]; }
        v16bf b = load_frag(Bbase, 256, kc);
        acc = __builtin_amdgcn_wmma_f32_16x16x32_bf16(
            false, a, false, b, (short)0, acc, false, false);
    }

    int nloc = lane & 15;
    int ncol = w * 16 + nloc;
    int moff = (lane & 16) ? 8 : 0;
    float bc = fcb[ncol];
#pragma unroll
    for (int r = 0; r < 8; ++r) {
        int mm = r + moff;
        out[(size_t)(e0 + mm) * 64 + ncol] = gelu_tanh(acc[r] + bc);
    }
}

// ---------------------------------------------------------------------------
// host launcher
// ---------------------------------------------------------------------------
static inline size_t align256(size_t x) { return (x + 255) & ~(size_t)255; }

extern "C" void kernel_launch(void* const* d_in, const int* in_sizes, int n_in,
                              void* d_out, int out_size, void* d_ws, size_t ws_size,
                              hipStream_t stream) {
    const int N = in_sizes[0] / 65;     // x is [N,65]
    const int E = in_sizes[1] / 2;      // edge_index is [2,E]

    const float* x   = (const float*)d_in[0];
    const int*   ei  = (const int*)d_in[1];
    const int*   row = ei;
    const int*   col = ei + E;
    const float* emb = (const float*)d_in[3];
    const float* W1  = (const float*)d_in[4];
    const float* b1  = (const float*)d_in[5];
    const float* W2  = (const float*)d_in[6];
    const float* b2  = (const float*)d_in[7];
    const float* W3  = (const float*)d_in[8];
    const float* b3  = (const float*)d_in[9];
    const float* fcW = (const float*)d_in[10];
    const float* fcb = (const float*)d_in[11];
    float* out = (float*)d_out;

    // scratch carving
    char* ws = (char*)d_ws;
    size_t off = 0;
    auto carve = [&](size_t bytes) -> void* {
        void* p = ws + off;
        off = align256(off + bytes);
        return p;
    };
    float*  dis  = (float*)carve((size_t)N * 4);          // deg -> dis in place
    bf16_t* Wt1  = (bf16_t*)carve(128 * 128 * 2);
    bf16_t* Wt2  = (bf16_t*)carve(128 * 128 * 2);
    bf16_t* Wt3  = (bf16_t*)carve(128 * 128 * 2);
    bf16_t* fcWt = (bf16_t*)carve(64 * 256 * 2);
    bf16_t* Hbf  = (bf16_t*)carve((size_t)N * 128 * 2);
    float*  Hc   = (float*)carve((size_t)N * 128 * 4);
    float*  agg  = (float*)carve((size_t)N * 128 * 4);

    const int T = 256;
    // weights -> transposed bf16
    transpose_to_bf16<<<(128 * 128 + T - 1) / T, T, 0, stream>>>(W1, Wt1, 128, 128);
    transpose_to_bf16<<<(128 * 128 + T - 1) / T, T, 0, stream>>>(W2, Wt2, 128, 128);
    transpose_to_bf16<<<(128 * 128 + T - 1) / T, T, 0, stream>>>(W3, Wt3, 128, 128);
    transpose_to_bf16<<<(256 * 64  + T - 1) / T, T, 0, stream>>>(fcW, fcWt, 256, 64);

    // degrees -> dis = deg^-1/2
    deg_init<<<(N + T - 1) / T, T, 0, stream>>>(dis, N);
    deg_accum<<<(E + T - 1) / T, T, 0, stream>>>(col, dis, E);
    deg_finalize<<<(N + T - 1) / T, T, 0, stream>>>(dis, N);

    // h0
    build_h0<<<((size_t)N * 128 + T - 1) / T, T, 0, stream>>>(x, emb, Hbf, N);

    const bf16_t* Wts[3]  = {Wt1, Wt2, Wt3};
    const float*  bias[3] = {b1, b2, b3};
    int mtiles  = N / 16;
    int sblocks = (int)(((size_t)E * 32 + T - 1) / T);
    int eblocks = (int)(((size_t)N * 128 + T - 1) / T);

    for (int layer = 0; layer < 3; ++layer) {
        gcn_gemm<<<mtiles, 256, 0, stream>>>(Hbf, Wts[layer], bias[layer], dis, Hc, agg);
        gcn_scatter<<<sblocks, T, 0, stream>>>(row, col, dis, Hc, agg, E);
        gelu_to_bf16<<<eblocks, T, 0, stream>>>(agg, Hbf, N * 128);
    }

    edge_mlp<<<E / 16, 128, 0, stream>>>(row, col, Hbf, fcWt, fcb, out, E);
}
